// OrthoLinear_18588618457624
// MI455X (gfx1250) — matrix operand order repaired and checked by reading, not compiled, not run on hardware
//
#include <hip/hip_runtime.h>

// ---------------------------------------------------------------------------
// OrthoLinear for MI455X (gfx1250, wave32):
//   out = X @ dequant_int4(W)^T + alpha * (X @ CSR^T)
//
// Roofline: dominant traffic is the int32-carrier weight stream (134 MB) ->
// ~5.8 us at 23.3 TB/s HBM. Dense compute (1.07 G MAC) rides on
// v_wmma_f32_16x16x32_f16 (int4 exact in f16, f32 accumulate). We therefore
// optimize for: (a) pure b128 streaming of weights, (b) minimal VALU decode
// per carrier (magic-constant nibble->f16 trick), (c) X pre-converted to f16
// once in d_ws so the A fragment needs zero in-loop conversion.
// ---------------------------------------------------------------------------

typedef __attribute__((ext_vector_type(16))) _Float16 v16h;
typedef __attribute__((ext_vector_type(8)))  _Float16 v8h;
typedef __attribute__((ext_vector_type(4)))  _Float16 v4h;
typedef __attribute__((ext_vector_type(2)))  _Float16 v2h;
typedef __attribute__((ext_vector_type(8)))  float    v8f;
typedef __attribute__((ext_vector_type(4)))  float    v4f;
typedef __attribute__((ext_vector_type(4)))  int      v4i;

#define TOKENS 16
#define IN_F   8192
#define OUT_F  8192
#define PACK   (IN_F / 2)        // int32 carriers per weight row (1 byte each)
#define KSPLIT 4
#define KSLICE (IN_F / KSPLIT)   // 2048

// Decode one int32 carrier (byte b = two packed int4) into a packed f16 pair
// (low nibble -> half 0, high nibble -> half 1), exactly:
//   t = ((e<<12)|e) & 0x000F000F        nibbles into both 16-bit mantissas
//   t ^= 0x64086408                     ^8 sign-offset + exponent of 1024.0
//   pair = bitcast<half2>(t) - 1032.0   -> ((nib^8)+1024) - 1032 = sext4(nib)
__device__ __forceinline__ v2h decode_pair(unsigned e) {
    unsigned t = ((e << 12) | e) & 0x000F000Fu;
    t ^= 0x64086408u;
    const v2h magic = __builtin_bit_cast(v2h, 0x64086408u);   // {1032h, 1032h}
    return __builtin_bit_cast(v2h, t) - magic;                // v_pk_sub_f16
}

// ---------------------------------------------------------------------------
// Pre-pass: X f32 -> f16 once (removes all A-side conversion from the hot
// loop and halves A vmem traffic). 131072 elements, 4 per thread.
// ---------------------------------------------------------------------------
__global__ __launch_bounds__(256) void x_to_f16(
    const float* __restrict__ x, _Float16* __restrict__ xh)
{
    const int i = (blockIdx.x * 256 + threadIdx.x) * 4;
    v4f v = *(const v4f*)(x + i);
    v4h h;
#pragma unroll
    for (int j = 0; j < 4; ++j) h[j] = (_Float16)v[j];
    *(v4h*)(xh + i) = h;
}

// ---------------------------------------------------------------------------
// Dense int4-dequant GEMM, f16 X path.
// grid = 256 blocks x 256 threads = 2048 waves; wave gw: tile = gw>>2
// (16 out-features), kslice = gw&3. One v_wmma_f32_16x16x32_f16 per K-step.
// ---------------------------------------------------------------------------
__global__ __launch_bounds__(256) void ortho_dense_wmma_h(
    const _Float16* __restrict__ xh,   // [16, IN_F] f16 (from pre-pass)
    const int*      __restrict__ bw,   // [OUT_F, PACK] int32 carriers
    const float*    __restrict__ scales,
    float*          __restrict__ out)  // [16, OUT_F] f32 (pre-zeroed)
{
    const int lane = threadIdx.x & 31;
    const int wave = threadIdx.x >> 5;
    const int gw   = blockIdx.x * 8 + wave;
    const int tile = gw >> 2;
    const int ksl  = gw & 3;
    const int n0   = tile * 16;

    const int mrow = lane & 15;
    const int hi   = lane >> 4;

    // A layout (ISA 16-bit A 16x32): lanes 0-15 row=lane, K {0..7,16..23};
    // lanes 16-31 row=lane-16, K {8..15,24..31}  -> two b128 half loads.
    const _Float16* xa = xh + mrow * IN_F + hi * 8;

    // B layout (ISA 16-bit B KxN): lanes 0-15 col=lane, K {0..7,16..23};
    // lanes 16-31 col=lane-16, K {8..15,24..31} -> 2x b128 of 4 carriers.
    const int* wb = bw + (n0 + mrow) * PACK + hi * 4;

    v8f acc = {};
    const int k_begin = ksl * KSLICE;
    const int k_end   = k_begin + KSLICE;

    for (int kk = k_begin; kk < k_end; kk += 32) {
        __builtin_prefetch(wb + ((kk + 32) >> 1), 0, 1);  // global_prefetch_b8

        // ---- A fragment: two b128 loads, no conversion
        v8h a0 = *(const v8h*)(xa + kk);
        v8h a1 = *(const v8h*)(xa + kk + 16);
        v16h A;
#pragma unroll
        for (int i = 0; i < 8; ++i) { A[i] = a0[i]; A[8 + i] = a1[i]; }

        // ---- B fragment: two b128 carrier loads + magic decode
        v4i w0 = *(const v4i*)(wb + (kk >> 1));
        v4i w1 = *(const v4i*)(wb + (kk >> 1) + 8);
        v16h B;
#pragma unroll
        for (int i = 0; i < 4; ++i) {
            v2h p0 = decode_pair((unsigned)w0[i]);
            v2h p1 = decode_pair((unsigned)w1[i]);
            B[2 * i + 0]     = p0[0];  B[2 * i + 1]     = p0[1];
            B[8 + 2 * i + 0] = p1[0];  B[8 + 2 * i + 1] = p1[1];
        }

        acc = __builtin_amdgcn_wmma_f32_16x16x32_f16(
            false, A, false, B, (short)0, acc, false, false);
    }

    // C/D layout: VGPR g -> (M = g + 8*hi, N = lane&15)
    const float s = scales[n0 + mrow];
#pragma unroll
    for (int g = 0; g < 8; ++g) {
        const int m = g + hi * 8;
        atomicAdd(&out[m * OUT_F + n0 + mrow], acc[g] * s);
    }
}

// ---------------------------------------------------------------------------
// Fallback dense kernel (no workspace): identical, but converts X f32->f16
// in-loop. Used only if ws_size is too small for the f16 X copy.
// ---------------------------------------------------------------------------
__global__ __launch_bounds__(256) void ortho_dense_wmma_f(
    const float* __restrict__ x,
    const int*   __restrict__ bw,
    const float* __restrict__ scales,
    float*       __restrict__ out)
{
    const int lane = threadIdx.x & 31;
    const int wave = threadIdx.x >> 5;
    const int gw   = blockIdx.x * 8 + wave;
    const int tile = gw >> 2;
    const int ksl  = gw & 3;
    const int n0   = tile * 16;

    const int mrow = lane & 15;
    const int hi   = lane >> 4;

    const float* xa = x + mrow * IN_F + hi * 8;
    const int*   wb = bw + (n0 + mrow) * PACK + hi * 4;

    v8f acc = {};
    const int k_begin = ksl * KSLICE;
    const int k_end   = k_begin + KSLICE;

    for (int kk = k_begin; kk < k_end; kk += 32) {
        __builtin_prefetch(wb + ((kk + 32) >> 1), 0, 1);

        v4f a0 = *(const v4f*)(xa + kk);
        v4f a1 = *(const v4f*)(xa + kk + 4);
        v4f a2 = *(const v4f*)(xa + kk + 16);
        v4f a3 = *(const v4f*)(xa + kk + 20);
        v16h A;
#pragma unroll
        for (int i = 0; i < 4; ++i) {
            A[i]      = (_Float16)a0[i];
            A[4 + i]  = (_Float16)a1[i];
            A[8 + i]  = (_Float16)a2[i];
            A[12 + i] = (_Float16)a3[i];
        }

        v4i w0 = *(const v4i*)(wb + (kk >> 1));
        v4i w1 = *(const v4i*)(wb + (kk >> 1) + 8);
        v16h B;
#pragma unroll
        for (int i = 0; i < 4; ++i) {
            v2h p0 = decode_pair((unsigned)w0[i]);
            v2h p1 = decode_pair((unsigned)w1[i]);
            B[2 * i + 0]     = p0[0];  B[2 * i + 1]     = p0[1];
            B[8 + 2 * i + 0] = p1[0];  B[8 + 2 * i + 1] = p1[1];
        }

        acc = __builtin_amdgcn_wmma_f32_16x16x32_f16(
            false, A, false, B, (short)0, acc, false, false);
    }

    const float s = scales[n0 + mrow];
#pragma unroll
    for (int g = 0; g < 8; ++g) {
        const int m = g + hi * 8;
        atomicAdd(&out[m * OUT_F + n0 + mrow], acc[g] * s);
    }
}

// ---------------------------------------------------------------------------
// CSR sparse branch: one wave per output row; lanes 0-15 = tokens, the two
// half-waves split the nnz list, combined with a lane-16 shuffle, one f32
// atomic per (token,row). X gathers hit L2 (X is 512 KB, L2 is 192 MB).
// ---------------------------------------------------------------------------
__global__ __launch_bounds__(128) void ortho_sparse_csr(
    const float* __restrict__ x,
    const float* __restrict__ vals,
    const int*   __restrict__ cols,
    const int*   __restrict__ rptr,
    const float* __restrict__ alpha,
    float*       __restrict__ out)
{
    const int row  = blockIdx.x * 4 + (threadIdx.x >> 5);
    const int lane = threadIdx.x & 31;
    const int tok  = lane & 15;
    const int half = lane >> 4;

    const int jb = rptr[row];
    const int je = rptr[row + 1];

    float acc = 0.0f;
    for (int j = jb + half; j < je; j += 2) {
        acc += x[tok * IN_F + cols[j]] * vals[j];
    }
    acc += __shfl_xor(acc, 16, 32);
    if (half == 0) {
        atomicAdd(&out[tok * OUT_F + row], alpha[0] * acc);
    }
}

// ---------------------------------------------------------------------------
// Host entry: memset -> (optional X f16 pre-pass) -> dense (atomic partials)
// -> sparse (atomic adds). All on `stream`; graph-capture safe; decision on
// ws_size is the same every call (deterministic).
// ---------------------------------------------------------------------------
extern "C" void kernel_launch(void* const* d_in, const int* in_sizes, int n_in,
                              void* d_out, int out_size, void* d_ws, size_t ws_size,
                              hipStream_t stream) {
    (void)in_sizes; (void)n_in;

    const float* x      = (const float*)d_in[0];
    const int*   bw     = (const int*)  d_in[1];
    const float* scales = (const float*)d_in[2];
    const float* vals   = (const float*)d_in[3];
    const int*   cols   = (const int*)  d_in[4];
    const int*   rptr   = (const int*)  d_in[5];
    const float* alpha  = (const float*)d_in[6];
    float*       out    = (float*)d_out;

    hipMemsetAsync(out, 0, (size_t)out_size * sizeof(float), stream);

    const size_t xh_bytes = (size_t)TOKENS * IN_F * sizeof(_Float16);  // 256 KB
    if (ws_size >= xh_bytes && d_ws != nullptr) {
        _Float16* xh = (_Float16*)d_ws;
        x_to_f16<<<(TOKENS * IN_F) / (256 * 4), 256, 0, stream>>>(x, xh);
        ortho_dense_wmma_h<<<256, 256, 0, stream>>>(xh, bw, scales, out);
    } else {
        ortho_dense_wmma_f<<<256, 256, 0, stream>>>(x, bw, scales, out);
    }

    ortho_sparse_csr<<<OUT_F / 4, 128, 0, stream>>>(x, vals, cols, rptr, alpha, out);
}
